// PositionPredictor_41120016892403
// MI455X (gfx1250) — matrix operand order repaired and checked by reading, not compiled, not run on hardware
//
#include <hip/hip_runtime.h>
#include <hip/hip_bf16.h>
#include <math.h>

// ---------------------------------------------------------------------------
// PositionPredictor fused kernel for gfx1250 (MI455X).
//   B=65536, K=32, H=512
//   Big GEMMs (ctx@W1^T, h@W2^T) run on v_wmma_f32_16x16x32_bf16 with
//   2x2 register blocking (2 row tiles x 2 N tiles per wave per K-step).
// ---------------------------------------------------------------------------

typedef __attribute__((ext_vector_type(16))) __bf16 v16bf;
typedef __attribute__((ext_vector_type(8)))  __bf16 v8bf;
typedef __attribute__((ext_vector_type(8)))  float  v8f;

#define HDIM  512
#define MTILE 32
#define SMEM_BYTES (2 * MTILE * HDIM * 2 + MTILE * 4 + MTILE * 3 * 4)  // 66048

// ---- weight prep: f32 (N,K) row-major -> bf16 (N,K) row-major ---------------
__global__ __launch_bounds__(256)
void pp_prep_weights(const float* __restrict__ W1, const float* __restrict__ W2,
                     const float* __restrict__ S1, const float* __restrict__ W3,
                     __bf16* __restrict__ Wt1, __bf16* __restrict__ Wt2,
                     __bf16* __restrict__ WtS1, __bf16* __restrict__ Wt3)
{
    int idx = blockIdx.x * 256 + threadIdx.x;     // 0 .. 512*512-1
    int k = idx >> 9;                             // K index 0..511
    int n = idx & 511;                            // N index 0..511
    Wt1[n * HDIM + k] = (__bf16)W1[n * HDIM + k]; // Wt1[n*512 + k] = W1[n][k]
    Wt2[n * HDIM + k] = (__bf16)W2[n * HDIM + k];
    if (n < 32) WtS1[n * HDIM + k] = (__bf16)S1[n * HDIM + k];
    if (n < 16) Wt3[n * HDIM + k]  = (n < 3) ? (__bf16)W3[n * HDIM + k] : (__bf16)0.0f;
}

// ---- A fragment loader: 16x32 bf16 tile from LDS ---------------------------
__device__ __forceinline__
v16bf load_afrag(const __bf16* __restrict__ Asrc, int lda, int m0, int k0,
                 int half, int mr)
{
    const __bf16* ap = Asrc + (m0 + mr) * lda + k0 + half * 8;
    v8bf lo = *(const v8bf*)(ap);
    v8bf hi = *(const v8bf*)(ap + 16);
    return __builtin_shufflevector(lo, hi, 0,1,2,3,4,5,6,7,8,9,10,11,12,13,14,15);
}

// ---- single 16x16 tile over K (for the small heads) ------------------------
__device__ __forceinline__
v8f wmma_tile(const __bf16* __restrict__ Asrc, int lda,
              const __bf16* __restrict__ Wt, int ldn,
              int m0, int n0, int Klen, int lane)
{
    const int half = lane >> 4;
    const int mr   = lane & 15;
    v8f acc = {0.f, 0.f, 0.f, 0.f, 0.f, 0.f, 0.f, 0.f};
#pragma unroll
    for (int k0 = 0; k0 < Klen; k0 += 32) {
        v16bf a = load_afrag(Asrc, lda, m0, k0, half, mr);
        v16bf b = load_afrag(Wt, HDIM, n0, k0, half, mr);
        acc = __builtin_amdgcn_wmma_f32_16x16x32_bf16(
                  false, a, false, b, (short)0, acc, false, false);
    }
    return acc;
}

// ---- 2x2 register-blocked layer: Dst = relu(Asrc @ Wt + bias) --------------
__device__ __forceinline__
void layer_gemm(const __bf16* __restrict__ Asrc,
                const __bf16* __restrict__ Wt,
                const float* __restrict__ bias,
                __bf16* __restrict__ Dst,
                int lane, int wv)
{
    const int half = lane >> 4;
    const int mr   = lane & 15;
    // 16 N-tile pairs (32 N-tiles), 2 row tiles each -> quads across 8 waves
    for (int qp = wv; qp < 16; qp += 8) {
        const int n0 = qp * 32;
        v8f acc00 = {0,0,0,0,0,0,0,0};
        v8f acc01 = {0,0,0,0,0,0,0,0};
        v8f acc10 = {0,0,0,0,0,0,0,0};
        v8f acc11 = {0,0,0,0,0,0,0,0};
#pragma unroll
        for (int k0 = 0; k0 < HDIM; k0 += 32) {
            v16bf a0 = load_afrag(Asrc, HDIM,  0, k0, half, mr);
            v16bf a1 = load_afrag(Asrc, HDIM, 16, k0, half, mr);
            const __bf16* wp = Wt + (size_t)(k0 + lane) * HDIM + n0;
            v16bf b0 = load_afrag(Wt, HDIM, n0, k0, half, mr);
            v16bf b1 = load_afrag(Wt, HDIM, n0 + 16, k0, half, mr);
            if (k0 + 32 < HDIM)
                __builtin_prefetch(wp + (size_t)32 * HDIM, 0, 1);
            acc00 = __builtin_amdgcn_wmma_f32_16x16x32_bf16(false, a0, false, b0, (short)0, acc00, false, false);
            acc01 = __builtin_amdgcn_wmma_f32_16x16x32_bf16(false, a0, false, b1, (short)0, acc01, false, false);
            acc10 = __builtin_amdgcn_wmma_f32_16x16x32_bf16(false, a1, false, b0, (short)0, acc10, false, false);
            acc11 = __builtin_amdgcn_wmma_f32_16x16x32_bf16(false, a1, false, b1, (short)0, acc11, false, false);
        }
#pragma unroll
        for (int r = 0; r < 8; ++r) {
            const int mrow = r + 8 * half;
            const int ncol = n0 + mr;
            float v00 = acc00[r] + bias[ncol];
            float v01 = acc01[r] + bias[ncol + 16];
            float v10 = acc10[r] + bias[ncol];
            float v11 = acc11[r] + bias[ncol + 16];
            Dst[(mrow     ) * HDIM + ncol     ] = (__bf16)(v00 > 0.f ? v00 : 0.f);
            Dst[(mrow     ) * HDIM + ncol + 16] = (__bf16)(v01 > 0.f ? v01 : 0.f);
            Dst[(mrow + 16) * HDIM + ncol     ] = (__bf16)(v10 > 0.f ? v10 : 0.f);
            Dst[(mrow + 16) * HDIM + ncol + 16] = (__bf16)(v11 > 0.f ? v11 : 0.f);
        }
    }
}

// ---- fused main kernel -----------------------------------------------------
__global__ __launch_bounds__(256)
void pp_main(const float* __restrict__ ctx, const float* __restrict__ partial,
             const float* __restrict__ b1, const float* __restrict__ b2,
             const float* __restrict__ b3, const float* __restrict__ bs1,
             const float* __restrict__ S2, const float* __restrict__ bs2,
             const __bf16* __restrict__ Wt1, const __bf16* __restrict__ Wt2,
             const __bf16* __restrict__ WtS1, const __bf16* __restrict__ Wt3,
             float* __restrict__ out)
{
    extern __shared__ char smem[];
    __bf16* As  = (__bf16*)smem;                              // 32 KB (ctx, then h2)
    __bf16* Hs  = (__bf16*)(smem + MTILE * HDIM * 2);         // 32 KB (h1; s overlay)
    float*  scl = (float*)(smem + 2 * MTILE * HDIM * 2);      // 32 floats
    float*  offs = scl + MTILE;                               // 96 floats
    float*  sS  = (float*)Hs;                                 // 32x32 f32 overlay

    const int tid  = threadIdx.x;
    const int lane = tid & 31;
    const int wv   = tid >> 5;
    const int half = lane >> 4;
    const int mr   = lane & 15;
    const int row0 = blockIdx.x * MTILE;

    // stage 0: context tile f32 -> bf16 LDS
    for (int i = tid; i < MTILE * HDIM; i += 256) {
        int r = i >> 9, c = i & 511;
        As[i] = (__bf16)ctx[(size_t)(row0 + r) * HDIM + c];
    }
    __syncthreads();

    // stage 1: scale branch  s = relu(ctx @ S1^T + bs1)   (2 row tiles x 2 N tiles)
    if (wv < 4) {
        const int rt = wv & 1, nt = wv >> 1;
        const int m0 = rt * 16, n0 = nt * 16;
        v8f acc = wmma_tile(As, HDIM, WtS1, 32, m0, n0, HDIM, lane);
#pragma unroll
        for (int r = 0; r < 8; ++r) {
            int m = m0 + r + 8 * half, n = n0 + mr;
            float v = acc[r] + bs1[n];
            sS[m * 32 + n] = v > 0.f ? v : 0.f;
        }
    }
    __syncthreads();

    // scale = softplus(s @ S2^T + bs2)
    if (tid < MTILE) {
        float x = bs2[0];
#pragma unroll
        for (int j = 0; j < 32; ++j) x += sS[tid * 32 + j] * S2[j];
        scl[tid] = (x > 20.f) ? x : log1pf(__expf(x));
    }
    __syncthreads();

    // stage 2: layer1  Hs = relu(As @ Wt1 + b1)
    layer_gemm(As, Wt1, b1, Hs, lane, wv);
    __syncthreads();

    // stage 3: layer2  As = relu(Hs @ Wt2 + b2)   (ctx dead, reuse As)
    layer_gemm(Hs, Wt2, b2, As, lane, wv);
    __syncthreads();

    // stage 4: offset = h2 @ W3^T + b3  (2 padded 16-col tiles, waves 0/1)
    if (wv < 2) {
        const int m0 = wv * 16;
        v8f acc = wmma_tile(As, HDIM, Wt3, 16, m0, 0, HDIM, lane);
#pragma unroll
        for (int r = 0; r < 8; ++r) {
            int m = m0 + r + 8 * half, n = mr;
            if (n < 3) offs[m * 3 + n] = acc[r] + b3[n];
        }
    }
    __syncthreads();

    // stage 5: per-row equivariant frame (centroid, cov, 3x3 Jacobi eig) + output
    if (tid < MTILE) {
        const float* P = partial + (size_t)(row0 + tid) * 96;   // 32 x 3
        float cx = 0.f, cy = 0.f, cz = 0.f;
        for (int k = 0; k < 32; ++k) { cx += P[3*k]; cy += P[3*k+1]; cz += P[3*k+2]; }
        cx *= (1.f/32.f); cy *= (1.f/32.f); cz *= (1.f/32.f);

        float A[3][3] = {{0,0,0},{0,0,0},{0,0,0}};
        for (int k = 0; k < 32; ++k) {
            float dx = P[3*k] - cx, dy = P[3*k+1] - cy, dz = P[3*k+2] - cz;
            A[0][0] += dx*dx; A[0][1] += dx*dy; A[0][2] += dx*dz;
            A[1][1] += dy*dy; A[1][2] += dy*dz; A[2][2] += dz*dz;
        }
        A[1][0] = A[0][1]; A[2][0] = A[0][2]; A[2][1] = A[1][2];

        float V[3][3] = {{1,0,0},{0,1,0},{0,0,1}};
        for (int sweep = 0; sweep < 8; ++sweep) {
            for (int pi = 0; pi < 3; ++pi) {
                const int p = (pi == 2) ? 1 : 0;
                const int q = (pi == 0) ? 1 : 2;
                float apq = A[p][q];
                if (fabsf(apq) < 1e-20f) continue;
                float theta = (A[q][q] - A[p][p]) / (2.f * apq);
                float t = copysignf(1.f, theta) / (fabsf(theta) + sqrtf(theta*theta + 1.f));
                float c = 1.f / sqrtf(t*t + 1.f);
                float s = t * c;
                for (int j = 0; j < 3; ++j) {           // rows
                    float ap = A[p][j], aq = A[q][j];
                    A[p][j] = c*ap - s*aq; A[q][j] = s*ap + c*aq;
                }
                for (int j = 0; j < 3; ++j) {           // cols (+ eigenvectors)
                    float ap = A[j][p], aq = A[j][q];
                    A[j][p] = c*ap - s*aq; A[j][q] = s*ap + c*aq;
                    float vp = V[j][p], vq = V[j][q];
                    V[j][p] = c*vp - s*vq; V[j][q] = s*vp + c*vq;
                }
            }
        }
        float ev[3] = {A[0][0], A[1][1], A[2][2]};
        for (int i = 0; i < 2; ++i)                     // sort eigenvalues desc
            for (int j = i + 1; j < 3; ++j)
                if (ev[j] > ev[i]) {
                    float te = ev[i]; ev[i] = ev[j]; ev[j] = te;
                    for (int r = 0; r < 3; ++r) { float tv = V[r][i]; V[r][i] = V[r][j]; V[r][j] = tv; }
                }

        float o0 = offs[tid*3], o1 = offs[tid*3+1], o2 = offs[tid*3+2];
        float sc = scl[tid];
        float cen[3] = {cx, cy, cz};
#pragma unroll
        for (int i = 0; i < 3; ++i) {
            // Vh = V^T  ->  global_offset[i] = sum_j V[j][i] * offset[j]
            float go = V[0][i]*o0 + V[1][i]*o1 + V[2][i]*o2;
            out[(size_t)(row0 + tid) * 3 + i] = cen[i] + go * sc;
        }
    }
}

// ---------------------------------------------------------------------------
extern "C" void kernel_launch(void* const* d_in, const int* in_sizes, int n_in,
                              void* d_out, int out_size, void* d_ws, size_t ws_size,
                              hipStream_t stream) {
    const float* ctx     = (const float*)d_in[0];   // (B, 512)
    const float* partial = (const float*)d_in[1];   // (B, 32, 3)
    const float* W1      = (const float*)d_in[2];   // (512, 512)
    const float* b1      = (const float*)d_in[3];
    const float* W2      = (const float*)d_in[4];   // (512, 512)
    const float* b2      = (const float*)d_in[5];
    const float* W3      = (const float*)d_in[6];   // (3, 512)
    const float* b3      = (const float*)d_in[7];
    const float* S1      = (const float*)d_in[8];   // (32, 512)
    const float* bs1     = (const float*)d_in[9];
    const float* S2      = (const float*)d_in[10];  // (1, 32)
    const float* bs2     = (const float*)d_in[11];
    float* out = (float*)d_out;

    const int B = in_sizes[0] / HDIM;               // 65536

    // workspace layout (bf16 weights, same (N,K) row-major layout as W)
    char* ws = (char*)d_ws;
    __bf16* Wt1  = (__bf16*)(ws);                        // 512*512*2 = 512 KB
    __bf16* Wt2  = (__bf16*)(ws + 524288);               // 512 KB
    __bf16* WtS1 = (__bf16*)(ws + 1048576);              // 512*32*2 = 32 KB
    __bf16* Wt3  = (__bf16*)(ws + 1048576 + 32768);      // 512*16*2 = 16 KB

    // allow 66 KB dynamic LDS (idempotent, host-side, capture-safe)
    (void)hipFuncSetAttribute((const void*)pp_main,
                              hipFuncAttributeMaxDynamicSharedMemorySize,
                              SMEM_BYTES);

    pp_prep_weights<<<(HDIM * HDIM) / 256, 256, 0, stream>>>(
        W1, W2, S1, W3, Wt1, Wt2, WtS1, Wt3);

    pp_main<<<B / MTILE, 256, SMEM_BYTES, stream>>>(
        ctx, partial, b1, b2, b3, bs1, S2, bs2, Wt1, Wt2, WtS1, Wt3, out);
}
